// GNN_mol_68891275428569
// MI455X (gfx1250) — compile-verified, matrix operand order checked by reading, not run on hardware
//
#include <hip/hip_runtime.h>
#include <hip/hip_bf16.h>
#include <math.h>

// ---------------------------------------------------------------------------
// GatedGCN-VN forward for MI455X (gfx1250).
// GEMMs (>95% of the 1.9 TFLOP forward) run on v_wmma_f32_16x16x32_bf16,
// wave32, 8 waves / 256-thread block, 128x128 block tile (32x64 per wave,
// 8 WMMAs per k-step), LDS staged with 18-bank (72B) padded strides so the
// packed bf16 stores are ds_store_b64 and fragment reads are conflict-free
// ds_load_2addr_b64 pairs. fp32->bf16 uses the native cast so the backend
// can pick the HW convert. Elementwise / scatter / BN passes are
// bandwidth-bound against the 23.3 TB/s HBM.
// ---------------------------------------------------------------------------

typedef __bf16 bf16_t;
typedef __attribute__((ext_vector_type(16))) __bf16 v16bf;
typedef __attribute__((ext_vector_type(8)))  float  v8f;

enum { NN = 60000, NE = 240000, NG = 2048, EMB = 256, HID = 1024,
       NTASK = 128, NLAYERS = 5 };

__device__ __forceinline__ bf16_t f2bf(float f) { return (bf16_t)f; }
__device__ __forceinline__ unsigned pack2bf(float a, float b) {
  const unsigned short ua = __builtin_bit_cast(unsigned short, (bf16_t)a);
  const unsigned short ub = __builtin_bit_cast(unsigned short, (bf16_t)b);
  return (unsigned)ua | ((unsigned)ub << 16);
}

// ----------------------------- WMMA GEMM -----------------------------------
// Y[M,N] = epilogue( X[M,K] @ W[K,N] + bias )  (+ resid)
// K multiple of 32, N multiple of 128. M arbitrary (edge blocks take a
// clamped-address path; interior blocks are branch-free and vectorized).
#define TILE_M 128
#define TILE_N 128
#define ASTR 36   // 72B row stride: 8B-aligned rows, 18-bank stride (conflict-free)
#define BSTR 36

__device__ __forceinline__ v16bf load_frag(const bf16_t* base, int lane, int ld) {
  const int half = lane >> 4;
  const int row  = lane & 15;
  v16bf f;
#pragma unroll
  for (int j = 0; j < 8; ++j) {
    // ISA 7.12.2: 16-bit A 16x32 layout. lanes0-15: K={0..7,16..23}; +8 for hi.
    const int kb = ((j < 4) ? (2 * j) : (16 + 2 * (j - 4))) + 8 * half;
    const bf16_t* p = base + row * ld + kb;
    f[2 * j]     = p[0];
    f[2 * j + 1] = p[1];
  }
  return f;
}

__global__ __launch_bounds__(256)
void gemm_bf16_wmma(const float* __restrict__ X, const float* __restrict__ W,
                    const float* __restrict__ bias, const float* __restrict__ resid,
                    float* __restrict__ Y, int M, int N, int K, int relu) {
  __shared__ bf16_t As[TILE_M * ASTR];   // 9216 B
  __shared__ bf16_t Bs[TILE_N * BSTR];   // 9216 B  (stored [n][k])

  const int tid  = threadIdx.x;
  const int lane = tid & 31;
  const int wid  = tid >> 5;
  const int wm   = wid >> 1;     // 4 waves along M (32 rows each)
  const int wn   = wid & 1;      // 2 waves along N (64 cols each)
  const int m0   = blockIdx.y * TILE_M;
  const int n0   = blockIdx.x * TILE_N;
  const bool full_m = (m0 + TILE_M <= M);

  v8f acc[2][4];
#pragma unroll
  for (int a = 0; a < 2; ++a)
#pragma unroll
    for (int b = 0; b < 4; ++b)
#pragma unroll
      for (int e = 0; e < 8; ++e) acc[a][b][e] = 0.0f;

  for (int k0 = 0; k0 < K; k0 += 32) {
    __syncthreads();
    if (full_m) {
      // ---- A tile 128x32 fp32, vectorized: 4 x float4 per thread ----
#pragma unroll
      for (int i = 0; i < 4; ++i) {
        const int idx = tid + i * 256;       // 0..1023 float4 slots
        const int r   = idx >> 3;            // row 0..127
        const int k4  = (idx & 7) * 4;       // k offset 0,4,..,28
        const float4 v = *(const float4*)(X + (size_t)(m0 + r) * K + (k0 + k4));
        uint2 pv; pv.x = pack2bf(v.x, v.y); pv.y = pack2bf(v.z, v.w);
        *(uint2*)(&As[r * ASTR + k4]) = pv;  // ds_store_b64
      }
      if (k0 + 32 < K)                        // overlap next slab with WMMA issue
        __builtin_prefetch(X + (size_t)(m0 + (tid >> 1)) * K + (k0 + 32), 0, 0);
    } else {
      // ---- edge blocks: clamp address, select 0 (no exec juggling) ----
#pragma unroll
      for (int i = 0; i < 8; ++i) {
        const int idx = tid + i * 256;       // 0..2047 bf16 pairs
        const int r  = idx >> 4;             // row 0..127
        const int kp = (idx & 15) * 2;       // k offset 0,2,..,30
        const int gr = m0 + r;
        const int cr = (gr < M) ? gr : (M - 1);
        const float* p = X + (size_t)cr * K + (k0 + kp);
        float v0 = p[0], v1 = p[1];
        if (gr >= M) { v0 = 0.0f; v1 = 0.0f; }
        *(unsigned*)(&As[r * ASTR + kp]) = pack2bf(v0, v1);
      }
    }
    // ---- B tile 32x128 fp32, vectorized loads, transposed [n][k] stores ----
#pragma unroll
    for (int i = 0; i < 4; ++i) {
      const int idx = tid + i * 256;         // 0..1023 float4 slots
      const int kk  = idx >> 5;              // 0..31
      const int n4  = (idx & 31) * 4;        // 0,4,..,124
      const float4 v = *(const float4*)(W + (size_t)(k0 + kk) * N + (n0 + n4));
      Bs[(n4 + 0) * BSTR + kk] = f2bf(v.x);
      Bs[(n4 + 1) * BSTR + kk] = f2bf(v.y);
      Bs[(n4 + 2) * BSTR + kk] = f2bf(v.z);
      Bs[(n4 + 3) * BSTR + kk] = f2bf(v.w);
    }
    __syncthreads();

    v16bf af[2], bfr[4];
#pragma unroll
    for (int t = 0; t < 2; ++t)
      af[t] = load_frag(&As[(wm * 32 + t * 16) * ASTR], lane, ASTR);
#pragma unroll
    for (int t = 0; t < 4; ++t)
      bfr[t] = load_frag(&Bs[(wn * 64 + t * 16) * BSTR], lane, BSTR);

#pragma unroll
    for (int a = 0; a < 2; ++a)
#pragma unroll
      for (int b = 0; b < 4; ++b)
        acc[a][b] = __builtin_amdgcn_wmma_f32_16x16x32_bf16(
            false, af[a], false, bfr[b], (short)0, acc[a][b], false, false);
  }

  // epilogue: C layout — lane n = lane&15; VGPR r -> M = r + 8*(lane>>4)
  const int half = lane >> 4;
  const int nl   = lane & 15;
#pragma unroll
  for (int a = 0; a < 2; ++a) {
#pragma unroll
    for (int b = 0; b < 4; ++b) {
      const int col = n0 + wn * 64 + b * 16 + nl;
#pragma unroll
      for (int r = 0; r < 8; ++r) {
        const int row = m0 + wm * 32 + a * 16 + half * 8 + r;
        if (row < M && col < N) {
          float v = acc[a][b][r];
          if (bias)  v += bias[col];
          if (relu)  v  = fmaxf(v, 0.0f);
          if (resid) v += resid[(size_t)row * N + col];
          Y[(size_t)row * N + col] = v;
        }
      }
    }
  }
}

// --------------------------- elementwise kernels ---------------------------

__global__ void node_embed_k(const int* __restrict__ af, const float* __restrict__ aemb,
                             const float* __restrict__ pe, const float* __restrict__ pw,
                             const float* __restrict__ pb, float* __restrict__ h) {
  size_t tid = (size_t)blockIdx.x * blockDim.x + threadIdx.x;
  if (tid >= (size_t)NN * EMB) return;
  const int n = (int)(tid >> 8), c = (int)(tid & 255);
  float s = pb[c];
#pragma unroll
  for (int f = 0; f < 9; ++f)
    s += aemb[((size_t)f * 100 + af[n * 9 + f]) * EMB + c];
#pragma unroll
  for (int d = 0; d < 10; ++d)
    s += pe[n * 10 + d] * pw[d * EMB + c];
  h[tid] = s;
}

__global__ void edge_embed_k(const int* __restrict__ bfeat, const float* __restrict__ bemb,
                             float* __restrict__ e) {
  size_t tid = (size_t)blockIdx.x * blockDim.x + threadIdx.x;
  if (tid >= (size_t)NE * EMB) return;
  const int ed = (int)(tid >> 8), c = (int)(tid & 255);
  float s = 0.0f;
#pragma unroll
  for (int f = 0; f < 3; ++f)
    s += bemb[((size_t)f * 10 + bfeat[ed * 3 + f]) * EMB + c];
  e[tid] = s;
}

__global__ void node_count_k(const int* __restrict__ ng, float* __restrict__ nc) {
  int t = blockIdx.x * blockDim.x + threadIdx.x;
  if (t < NN) atomicAdd(&nc[ng[t]], 1.0f);
}

__global__ void edge_graph_k(const int* __restrict__ src, const int* __restrict__ ng,
                             int* __restrict__ eg, float* __restrict__ ec) {
  int t = blockIdx.x * blockDim.x + threadIdx.x;
  if (t >= NE) return;
  const int g = ng[src[t]];
  eg[t] = g;
  atomicAdd(&ec[g], 1.0f);
}

__global__ void broadcast_row_k(const float* __restrict__ row, float* __restrict__ out) {
  size_t tid = (size_t)blockIdx.x * blockDim.x + threadIdx.x;
  if (tid < (size_t)NG * EMB) out[tid] = row[tid & 255];
}

__global__ void vn_add_k(float* __restrict__ h, const float* __restrict__ vn,
                         const int* __restrict__ ng) {
  size_t tid = (size_t)blockIdx.x * blockDim.x + threadIdx.x;
  if (tid >= (size_t)NN * EMB) return;
  const int n = (int)(tid >> 8), c = (int)(tid & 255);
  h[tid] += vn[(size_t)ng[n] * EMB + c];
}

__global__ void edge_hat_scatter_k(const float* __restrict__ Dh, const float* __restrict__ Eh,
                                   const float* __restrict__ Ce, const float* __restrict__ Bh,
                                   const int* __restrict__ src, const int* __restrict__ dst,
                                   float* __restrict__ ehat, float* __restrict__ num,
                                   float* __restrict__ den) {
  size_t tid = (size_t)blockIdx.x * blockDim.x + threadIdx.x;
  if (tid >= (size_t)NE * EMB) return;
  const int ed = (int)(tid >> 8), c = (int)(tid & 255);
  const int s = src[ed], d = dst[ed];
  const float eh = Dh[(size_t)s * EMB + c] + Eh[(size_t)d * EMB + c] + Ce[tid];
  ehat[tid] = eh;
  const float sg = 1.0f / (1.0f + __expf(-eh));
  atomicAdd(&num[(size_t)d * EMB + c], sg * Bh[(size_t)s * EMB + c]);
  atomicAdd(&den[(size_t)d * EMB + c], sg);
}

__global__ void node_gate_k(const float* __restrict__ Ah, const float* __restrict__ num,
                            const float* __restrict__ den, float* __restrict__ out) {
  size_t tid = (size_t)blockIdx.x * blockDim.x + threadIdx.x;
  if (tid < (size_t)NN * EMB) out[tid] = Ah[tid] + num[tid] / (den[tid] + 1e-6f);
}

// two-stage training-mode batchnorm column statistics
__global__ void col_stats_partial_k(const float* __restrict__ X, int M, int N,
                                    float* __restrict__ sumb, float* __restrict__ sqb) {
  __shared__ float ss[256], sq2[256];
  const int tx = threadIdx.x & 31, ty = threadIdx.x >> 5;
  const int c = blockIdx.x * 32 + tx;
  float s = 0.0f, q = 0.0f;
  for (int r = blockIdx.y * 8 + ty; r < M; r += gridDim.y * 8) {
    const float v = X[(size_t)r * N + c];
    s += v; q += v * v;
  }
  ss[threadIdx.x] = s; sq2[threadIdx.x] = q;
  __syncthreads();
  if (ty == 0) {
#pragma unroll
    for (int i = 1; i < 8; ++i) { s += ss[i * 32 + tx]; q += sq2[i * 32 + tx]; }
    atomicAdd(&sumb[c], s);
    atomicAdd(&sqb[c], q);
  }
}

__global__ void col_stats_final_k(const float* __restrict__ sumb, const float* __restrict__ sqb,
                                  int M, int N, float* __restrict__ mean, float* __restrict__ rstd) {
  const int c = blockIdx.x * blockDim.x + threadIdx.x;
  if (c >= N) return;
  const float mu = sumb[c] / (float)M;
  const float var = sqb[c] / (float)M - mu * mu;
  mean[c] = mu;
  rstd[c] = rsqrtf(fmaxf(var, 0.0f) + 1e-5f);
}

__global__ void bn_apply_k(const float* __restrict__ X, const float* __restrict__ mean,
                           const float* __restrict__ rstd, const float* __restrict__ g,
                           const float* __restrict__ b, const float* __restrict__ resid,
                           float* __restrict__ Y, size_t total, int relu) {
  size_t tid = (size_t)blockIdx.x * blockDim.x + threadIdx.x;
  if (tid >= total) return;
  const int c = (int)(tid & 255);
  float v = (X[tid] - mean[c]) * rstd[c] * g[c] + b[c];
  if (relu) v = fmaxf(v, 0.0f);
  Y[tid] = (resid ? resid[tid] : 0.0f) + v;
}

__global__ void seg_sum_k(const float* __restrict__ X, const int* __restrict__ seg,
                          float* __restrict__ out, size_t total) {
  size_t tid = (size_t)blockIdx.x * blockDim.x + threadIdx.x;
  if (tid >= total) return;
  const int m = (int)(tid >> 8), c = (int)(tid & 255);
  atomicAdd(&out[(size_t)seg[m] * EMB + c], X[tid]);
}

__global__ void vn_combine_k(const float* __restrict__ vn, const float* __restrict__ vsum,
                             const float* __restrict__ nc, float* __restrict__ out) {
  size_t tid = (size_t)blockIdx.x * blockDim.x + threadIdx.x;
  if (tid >= (size_t)NG * EMB) return;
  const int g = (int)(tid >> 8);
  out[tid] = vn[tid] + vsum[tid] / fmaxf(nc[g], 1.0f);
}

__global__ void concat_div_k(const float* __restrict__ hs, const float* __restrict__ es,
                             const float* __restrict__ nc, const float* __restrict__ ec,
                             float* __restrict__ hg) {
  size_t tid = (size_t)blockIdx.x * blockDim.x + threadIdx.x;
  if (tid >= (size_t)NG * 2 * EMB) return;
  const int g = (int)(tid >> 9), c = (int)(tid & 511);
  hg[tid] = (c < EMB) ? hs[(size_t)g * EMB + c] / fmaxf(nc[g], 1.0f)
                      : es[(size_t)g * EMB + (c - EMB)] / fmaxf(ec[g], 1.0f);
}

// ------------------------------- host side ---------------------------------

static inline dim3 g1(size_t n) { return dim3((unsigned)((n + 255) / 256)); }

extern "C" void kernel_launch(void* const* d_in, const int* in_sizes, int n_in,
                              void* d_out, int out_size, void* d_ws, size_t ws_size,
                              hipStream_t stream) {
  (void)in_sizes; (void)n_in; (void)out_size; (void)ws_size;
  auto F = [&](int i) { return (const float*)d_in[i]; };
  auto I = [&](int i) { return (const int*)d_in[i]; };

  // top-level inputs (insertion order); params pytree flattened with sorted keys
  const int*   atom_feat = I(0);
  const int*   bond_feat = I(1);
  const float* pos_enc   = F(2);
  const int*   src       = I(3);
  const int*   dst       = I(4);
  const int*   ng        = I(5);
  const float* atom_emb  = F(6);
  const float* bond_emb  = F(7);
  const int L0 = 8;                       // layers: 26 leaves each
  const float* pos_b  = F(138); const float* pos_w  = F(139);
  const float* p1b    = F(140); const float* p1w    = F(141);
  const float* p2b    = F(142); const float* p2w    = F(143);
  const float* vn_emb = F(144);
  const int V0 = 145;                     // vn_ff: bn.b, bn.g, l1.b, l1.w, l2.b, l2.w

  // -------- workspace carve --------
  char* base = (char*)d_ws; size_t off = 0;
  auto alloc = [&](size_t nf) {
    float* p = (float*)(base + off);
    off += ((nf * sizeof(float) + 255) / 256) * 256;
    return p;
  };
  float* h    = alloc((size_t)NN * EMB);
  float* e    = alloc((size_t)NE * EMB);
  float* Ah   = alloc((size_t)NN * EMB);
  float* Bh   = alloc((size_t)NN * EMB);
  float* Dh   = alloc((size_t)NN * EMB);
  float* Eh   = alloc((size_t)NN * EMB);
  float* Ce   = alloc((size_t)NE * EMB);
  float* ehat = alloc((size_t)NE * EMB);   // also reused as edge FFN pre-BN
  float* num  = alloc((size_t)NN * EMB);
  float* den  = alloc((size_t)NN * EMB);
  float* tmph = alloc((size_t)NN * EMB);
  float* hm   = alloc((size_t)NN * EMB);
  float* em   = alloc((size_t)NE * EMB);
  float* nhid = alloc((size_t)NN * HID);
  float* ehid = alloc((size_t)NE * HID);
  float* vn   = alloc((size_t)NG * EMB);
  float* vsum = alloc((size_t)NG * EMB);
  float* vin  = alloc((size_t)NG * EMB);
  float* vbn  = alloc((size_t)NG * EMB);
  float* vhid = alloc((size_t)NG * HID);
  float* hsum = alloc((size_t)NG * EMB);
  float* esum = alloc((size_t)NG * EMB);
  float* hg   = alloc((size_t)NG * 2 * EMB);
  float* ghid = alloc((size_t)NG * HID);
  float* ncnt = alloc(NG);
  float* ecnt = alloc(NG);
  float* mean = alloc(1024);
  float* rstd = alloc(1024);
  float* ssum = alloc(1024);
  float* ssq  = alloc(1024);
  int*   eg   = (int*)alloc(NE);

  auto gemm = [&](const float* X, const float* W, const float* b, const float* res,
                  float* Y, int M, int N, int K, int relu) {
    dim3 grid((N + TILE_N - 1) / TILE_N, (M + TILE_M - 1) / TILE_M);
    gemm_bf16_wmma<<<grid, 256, 0, stream>>>(X, W, b, res, Y, M, N, K, relu);
  };
  auto bn_stats = [&](const float* X, int M, int N) {
    hipMemsetAsync(ssum, 0, N * sizeof(float), stream);
    hipMemsetAsync(ssq,  0, N * sizeof(float), stream);
    col_stats_partial_k<<<dim3(N / 32, 64), 256, 0, stream>>>(X, M, N, ssum, ssq);
    col_stats_final_k<<<dim3((N + 255) / 256), 256, 0, stream>>>(ssum, ssq, M, N, mean, rstd);
  };

  const size_t NhE = (size_t)NN * EMB, NeE = (size_t)NE * EMB;
  const size_t NgE = (size_t)NG * EMB;

  // -------- encoders / counts / virtual node init --------
  node_embed_k<<<g1(NhE), 256, 0, stream>>>(atom_feat, atom_emb, pos_enc, pos_w, pos_b, h);
  edge_embed_k<<<g1(NeE), 256, 0, stream>>>(bond_feat, bond_emb, e);
  hipMemsetAsync(ncnt, 0, NG * sizeof(float), stream);
  hipMemsetAsync(ecnt, 0, NG * sizeof(float), stream);
  node_count_k<<<g1(NN), 256, 0, stream>>>(ng, ncnt);
  edge_graph_k<<<g1(NE), 256, 0, stream>>>(src, ng, eg, ecnt);
  broadcast_row_k<<<g1(NgE), 256, 0, stream>>>(vn_emb, vn);

  for (int li = 0; li < NLAYERS; ++li) {
    const int lb = L0 + 26 * li;
    const float *Ab_ = F(lb+0),  *Aw_ = F(lb+1);
    const float *Bb_ = F(lb+2),  *Bw_ = F(lb+3);
    const float *Cb_ = F(lb+4),  *Cw_ = F(lb+5);
    const float *Db_ = F(lb+6),  *Dw_ = F(lb+7);
    const float *Eb_ = F(lb+8),  *Ew_ = F(lb+9);
    const float *bne_b  = F(lb+10), *bne_g  = F(lb+11);
    const float *bne2_b = F(lb+12), *bne2_g = F(lb+13);
    const float *bnh_b  = F(lb+14), *bnh_g  = F(lb+15);
    const float *bnh2_b = F(lb+16), *bnh2_g = F(lb+17);
    const float *fe1b = F(lb+18), *fe1w = F(lb+19);
    const float *fe2b = F(lb+20), *fe2w = F(lb+21);
    const float *fh1b = F(lb+22), *fh1w = F(lb+23);
    const float *fh2b = F(lb+24), *fh2w = F(lb+25);

    // virtual-node broadcast add
    vn_add_k<<<g1(NhE), 256, 0, stream>>>(h, vn, ng);

    // gated-GCN linears (WMMA)
    gemm(h, Aw_, Ab_, nullptr, Ah, NN, EMB, EMB, 0);
    gemm(h, Bw_, Bb_, nullptr, Bh, NN, EMB, EMB, 0);
    gemm(h, Dw_, Db_, nullptr, Dh, NN, EMB, EMB, 0);
    gemm(h, Ew_, Eb_, nullptr, Eh, NN, EMB, EMB, 0);
    gemm(e, Cw_, Cb_, nullptr, Ce, NE, EMB, EMB, 0);

    // message passing (edge gather + atomic scatter)
    hipMemsetAsync(num, 0, NhE * sizeof(float), stream);
    hipMemsetAsync(den, 0, NhE * sizeof(float), stream);
    edge_hat_scatter_k<<<g1(NeE), 256, 0, stream>>>(Dh, Eh, Ce, Bh, src, dst, ehat, num, den);
    node_gate_k<<<g1(NhE), 256, 0, stream>>>(Ah, num, den, tmph);

    // h_mp = h + relu(bn_h(tmph)); e_mp = e + relu(bn_e(ehat))
    bn_stats(tmph, NN, EMB);
    bn_apply_k<<<g1(NhE), 256, 0, stream>>>(tmph, mean, rstd, bnh_g, bnh_b, h, hm, NhE, 1);
    bn_stats(ehat, NE, EMB);
    bn_apply_k<<<g1(NeE), 256, 0, stream>>>(ehat, mean, rstd, bne_g, bne_b, e, em, NeE, 1);

    // node FFN: h = bn_h2(hm + W2 relu(W1 hm))
    gemm(hm, fh1w, fh1b, nullptr, nhid, NN, HID, EMB, 1);
    gemm(nhid, fh2w, fh2b, hm, tmph, NN, EMB, HID, 0);
    bn_stats(tmph, NN, EMB);
    bn_apply_k<<<g1(NhE), 256, 0, stream>>>(tmph, mean, rstd, bnh2_g, bnh2_b, nullptr, h, NhE, 0);

    // edge FFN: e = bn_e2(em + W2 relu(W1 em))
    gemm(em, fe1w, fe1b, nullptr, ehid, NE, HID, EMB, 1);
    gemm(ehid, fe2w, fe2b, em, ehat, NE, EMB, HID, 0);
    bn_stats(ehat, NE, EMB);
    bn_apply_k<<<g1(NeE), 256, 0, stream>>>(ehat, mean, rstd, bne2_g, bne2_b, nullptr, e, NeE, 0);

    // virtual-node update
    if (li < NLAYERS - 1) {
      const int vb = V0 + 6 * li;
      const float *vbn_b = F(vb+0), *vbn_g = F(vb+1);
      const float *l1b = F(vb+2), *l1w = F(vb+3);
      const float *l2b = F(vb+4), *l2w = F(vb+5);
      hipMemsetAsync(vsum, 0, NgE * sizeof(float), stream);
      seg_sum_k<<<g1(NhE), 256, 0, stream>>>(h, ng, vsum, NhE);
      vn_combine_k<<<g1(NgE), 256, 0, stream>>>(vn, vsum, ncnt, vin);
      bn_stats(vin, NG, EMB);
      bn_apply_k<<<g1(NgE), 256, 0, stream>>>(vin, mean, rstd, vbn_g, vbn_b, nullptr, vbn, NgE, 0);
      gemm(vbn, l1w, l1b, nullptr, vhid, NG, HID, EMB, 1);
      gemm(vhid, l2w, l2b, nullptr, vn, NG, EMB, HID, 0);
    }
  }

  // -------- readout + prediction head --------
  hipMemsetAsync(hsum, 0, NgE * sizeof(float), stream);
  hipMemsetAsync(esum, 0, NgE * sizeof(float), stream);
  seg_sum_k<<<g1(NhE), 256, 0, stream>>>(h, ng, hsum, NhE);
  seg_sum_k<<<g1(NeE), 256, 0, stream>>>(e, eg, esum, NeE);
  concat_div_k<<<g1((size_t)NG * 2 * EMB), 256, 0, stream>>>(hsum, esum, ncnt, ecnt, hg);
  gemm(hg,   p1w, p1b, nullptr, ghid, NG, HID, 2 * EMB, 1);
  gemm(ghid, p2w, p2b, nullptr, (float*)d_out, NG, NTASK, HID, 0);
}
